// Net_57269093925096
// MI455X (gfx1250) — matrix operand order repaired and checked by reading, not compiled, not run on hardware
//
#include <hip/hip_runtime.h>

#define N_CLIENTS 100
#define NPAD      112          // 7 x 16 tiles
#define NTILES    49           // 7*7
#define NDROP     21           // n - (k+1) = 100 - 79
#define KSEL      79           // k+1
#define CHUNK     32           // K-slice per WMMA pass
#define COLSTRIDE 40           // padded ushort stride per column (80B = 20 banks, conflict-free)
#define GRAM_BLOCKS 512

typedef __attribute__((ext_vector_type(16))) __bf16         v16bf;
typedef __attribute__((ext_vector_type(8)))  float          v8f;
typedef __attribute__((ext_vector_type(8)))  unsigned short us8;

union ABop { us8 h[2]; v16bf v; };

__device__ __forceinline__ unsigned short f2bf(float f) {
  unsigned int u = __float_as_uint(f);
  u += 0x7FFFu + ((u >> 16) & 1u);        // round-to-nearest-even
  return (unsigned short)(u >> 16);
}
__device__ __forceinline__ float bf2f(unsigned short h) {
  return __uint_as_float(((unsigned int)h) << 16);
}

__global__ void zero_kernel(float* __restrict__ g, int n) {
  int i = blockIdx.x * blockDim.x + threadIdx.x;
  if (i < n) g[i] = 0.0f;
}

// G[i][j] = sum_d X[d][i]*X[d][j], X = (D x 100) row-major fp32.
// bf16 hi/lo split: x = hi + lo;  x_i*x_j ~= hi_i*hi_j + hi_i*lo_j + lo_i*hi_j
__global__ void __launch_bounds__(256)
gram_kernel(const float* __restrict__ in, float* __restrict__ G, int D) {
  __shared__ __align__(16) unsigned short lds_hi[NPAD * COLSTRIDE];
  __shared__ __align__(16) unsigned short lds_lo[NPAD * COLSTRIDE];

  const int tid  = threadIdx.x;
  const int wave = (tid >> 5) & 7;
  const int lane = tid & 31;
  const int half = lane >> 4;     // 0: lanes 0-15, 1: lanes 16-31
  const int l16  = lane & 15;
  const int chunks = (D + CHUNK - 1) / CHUNK;

  // pad columns 100..111 are constant zero; region is contiguous in LDS
  for (int f = tid; f < (NPAD - N_CLIENTS) * COLSTRIDE; f += 256) {
    lds_hi[N_CLIENTS * COLSTRIDE + f] = 0;
    lds_lo[N_CLIENTS * COLSTRIDE + f] = 0;
  }

  v8f acc[7];
#pragma unroll
  for (int s = 0; s < 7; ++s)
#pragma unroll
    for (int e = 0; e < 8; ++e) acc[s][e] = 0.0f;

  for (int chunk = blockIdx.x; chunk < chunks; chunk += gridDim.x) {
    // ---- stage CHUNK rows x N_CLIENTS cols as transposed bf16 hi/lo planes ----
    for (int f = tid; f < CHUNK * N_CLIENTS; f += 256) {
      int kk  = f / N_CLIENTS;
      int c   = f - kk * N_CLIENTS;
      int row = chunk * CHUNK + kk;
      float g = 0.0f;
      if (row < D) {
        const float* p = in + (size_t)row * N_CLIENTS + c;
        g = *p;
        if (chunk + (int)gridDim.x < chunks)   // stream this block's next chunk into L2
          __builtin_prefetch(p + (size_t)gridDim.x * CHUNK * N_CLIENTS, 0, 0);
      }
      unsigned short h = f2bf(g);
      unsigned short l = f2bf(g - bf2f(h));
      lds_hi[c * COLSTRIDE + kk] = h;
      lds_lo[c * COLSTRIDE + kk] = l;
    }
    __syncthreads();

    // ---- each wave: up to 7 of the 49 output tiles (static acc indices!) ----
#pragma unroll
    for (int s = 0; s < 7; ++s) {
      const int t = wave + 8 * s;
      if (t < NTILES) {                       // wave-uniform; false only for s==6, wave>0
        const int i0   = (t / 7) * 16;
        const int j0   = (t % 7) * 16;
        const int colA = i0 + l16;            // A: M = lane%16
        const int colB = j0 + l16;            // B: N = lane%16
        ABop ahi, alo, bhi, blo;
        // A (16x32, MxK): elems 0..7 -> K = 8*half.. ; 8..15 -> K = 16+8*half..
        ahi.h[0] = *(const us8*)&lds_hi[colA * COLSTRIDE + 8 * half];
        ahi.h[1] = *(const us8*)&lds_hi[colA * COLSTRIDE + 16 + 8 * half];
        alo.h[0] = *(const us8*)&lds_lo[colA * COLSTRIDE + 8 * half];
        alo.h[1] = *(const us8*)&lds_lo[colA * COLSTRIDE + 16 + 8 * half];
        // B (32x16, KxN): elems 0..7 -> K = 8*half.. ; 8..15 -> K = 16+8*half..
        bhi.h[0] = *(const us8*)&lds_hi[colB * COLSTRIDE + 8 * half];
        bhi.h[1] = *(const us8*)&lds_hi[colB * COLSTRIDE + 16 + 8 * half];
        blo.h[0] = *(const us8*)&lds_lo[colB * COLSTRIDE + 8 * half];
        blo.h[1] = *(const us8*)&lds_lo[colB * COLSTRIDE + 16 + 8 * half];

        acc[s] = __builtin_amdgcn_wmma_f32_16x16x32_bf16(false, ahi.v, false, bhi.v,
                                                         (short)0, acc[s], false, false);
        acc[s] = __builtin_amdgcn_wmma_f32_16x16x32_bf16(false, ahi.v, false, blo.v,
                                                         (short)0, acc[s], false, false);
        acc[s] = __builtin_amdgcn_wmma_f32_16x16x32_bf16(false, alo.v, false, bhi.v,
                                                         (short)0, acc[s], false, false);
      }
    }
    __syncthreads();
  }

  // ---- flush partial Gram tiles (C layout: VGPR r, lane L -> M=r+8*(L/16), N=L%16) ----
#pragma unroll
  for (int s = 0; s < 7; ++s) {
    const int t = wave + 8 * s;
    if (t < NTILES) {
      const int i0 = (t / 7) * 16;
      const int j0 = (t % 7) * 16;
      const int gj = j0 + l16;
#pragma unroll
      for (int r = 0; r < 8; ++r) {
        int gi = i0 + r + 8 * half;
        atomicAdd(&G[gi * NPAD + gj], acc[s][r]);
      }
    }
  }
}

// distances -> per-row sum of 79 smallest -> argmin -> 112-entry weight vector (1/79 or 0)
__global__ void __launch_bounds__(128)
select_kernel(const float* __restrict__ G, float* __restrict__ w) {
  __shared__ float dist[N_CLIENTS * N_CLIENTS];
  __shared__ float rowsum[N_CLIENTS];
  __shared__ int   istar_s;
  const int tid = threadIdx.x;

  for (int f = tid; f < N_CLIENTS * N_CLIENTS; f += 128) {
    int i = f / N_CLIENTS, j = f - i * N_CLIENTS;
    float d2 = G[i * NPAD + i] + G[j * NPAD + j] - 2.0f * G[i * NPAD + j];
    dist[f] = sqrtf(fmaxf(d2, 0.0f));
  }
  __syncthreads();

  if (tid < N_CLIENTS) {
    float top[NDROP];                       // NDROP largest (to subtract)
    for (int t = 0; t < NDROP; ++t) top[t] = -1.0f;
    float total = 0.0f;
    for (int j = 0; j < N_CLIENTS; ++j) {
      float v = dist[tid * N_CLIENTS + j];
      total += v;
      int   mi = 0; float mv = top[0];
      for (int t = 1; t < NDROP; ++t) if (top[t] < mv) { mv = top[t]; mi = t; }
      if (v > mv) top[mi] = v;
    }
    float drop = 0.0f;
    for (int t = 0; t < NDROP; ++t) drop += top[t];
    rowsum[tid] = total - drop;
  }
  __syncthreads();

  if (tid == 0) {
    int best = 0; float bv = rowsum[0];
    for (int i = 1; i < N_CLIENTS; ++i)
      if (rowsum[i] < bv) { bv = rowsum[i]; best = i; }
    istar_s = best;
  }
  __syncthreads();

  if (tid < NPAD) w[tid] = 0.0f;
  __syncthreads();

  if (tid == 0) {
    const int is = istar_s;
    bool picked[N_CLIENTS];
    for (int j = 0; j < N_CLIENTS; ++j) picked[j] = false;
    const float inv = 1.0f / (float)KSEL;
    for (int t = 0; t < KSEL; ++t) {        // 79 smallest of row i_star (incl. self, d=0)
      int bj = 0; float bv = 3.4e38f;
      for (int j = 0; j < N_CLIENTS; ++j)
        if (!picked[j] && dist[is * N_CLIENTS + j] < bv) { bv = dist[is * N_CLIENTS + j]; bj = j; }
      picked[bj] = true;
      w[bj] = inv;
    }
  }
}

// out[dd] = row(dd) . w   (w already scaled by 1/79)
__global__ void __launch_bounds__(64)
gather_kernel(const float* __restrict__ in, const float* __restrict__ w,
              float* __restrict__ out, int D) {
  __shared__ float ws[NPAD];
  const int tid = threadIdx.x;
  for (int c = tid; c < NPAD; c += 64) ws[c] = w[c];
  __syncthreads();

  long dd = (long)blockIdx.x * 64 + tid;
  if (dd < D) {
    const float4* row = (const float4*)(in + (size_t)dd * N_CLIENTS);  // 400B rows, 16B aligned
    float s = 0.0f;
#pragma unroll
    for (int q = 0; q < N_CLIENTS / 4; ++q) {
      float4 v = row[q];
      int c = q * 4;
      s += v.x * ws[c] + v.y * ws[c + 1] + v.z * ws[c + 2] + v.w * ws[c + 3];
    }
    out[dd] = s;
  }
}

extern "C" void kernel_launch(void* const* d_in, const int* in_sizes, int n_in,
                              void* d_out, int out_size, void* d_ws, size_t ws_size,
                              hipStream_t stream) {
  const float* in  = (const float*)d_in[0];
  float*       out = (float*)d_out;
  float*       G   = (float*)d_ws;              // 112*112 fp32 Gram accumulator
  float*       w   = G + NPAD * NPAD;           // 112 fp32 selection weights
  const int    D   = in_sizes[0] / N_CLIENTS;   // 1,000,000

  zero_kernel<<<(NPAD * NPAD + 255) / 256, 256, 0, stream>>>(G, NPAD * NPAD);
  gram_kernel<<<GRAM_BLOCKS, 256, 0, stream>>>(in, G, D);
  select_kernel<<<1, 128, 0, stream>>>(G, w);
  gather_kernel<<<(D + 63) / 64, 64, 0, stream>>>(in, w, out, D);
}
